// Discriminator_35201551958046
// MI455X (gfx1250) — matrix operand order repaired and checked by reading, not compiled, not run on hardware
//
#include <hip/hip_runtime.h>

// CDNA5 / gfx1250, wave32.
typedef __bf16 bf16_t;
typedef __attribute__((ext_vector_type(16))) __bf16 v16bf;
typedef __attribute__((ext_vector_type(8)))  float  v8f;
typedef __attribute__((ext_vector_type(4)))  __bf16 v4bf;

#define DDIM 1024
#define BATCH 64
#define NEG_SLOPE 0.01f

struct b32x8 { uint4 a, b; };   // 32-byte blob for bit_cast into v16bf

// ---------------------------------------------------------------------------
// Kernel A: Bsum[b,i] = sum_j |x[b,i] - x[b,j]|
// ---------------------------------------------------------------------------
__global__ __launch_bounds__(256) void bsum_kernel(const float* __restrict__ x,
                                                   float* __restrict__ bsum) {
    __shared__ float sx[DDIM];
    const int b = blockIdx.y;
    const int tid = threadIdx.x;
    for (int j = tid; j < DDIM; j += 256) sx[j] = x[b * DDIM + j];
    __syncthreads();
    const int i = blockIdx.x * 256 + tid;
    const float si = sx[i];
    float acc = 0.f;
    #pragma unroll 8
    for (int j = 0; j < DDIM; ++j) acc += fabsf(si - sx[j]);
    bsum[b * DDIM + i] = acc;
}

// ---------------------------------------------------------------------------
// Kernel B: fused neural-sort softmax + contraction (no [B,D,D] tensor).
//   xs[b,i] = sum_j softmax_j((D-1-2i)*x_j - Bsum_j) * x_j   -> bf16
// One wave32 per output row; x/Bsum rows staged in LDS.
// ---------------------------------------------------------------------------
__global__ __launch_bounds__(256) void softsort_kernel(const float* __restrict__ x,
                                                       const float* __restrict__ bsum,
                                                       bf16_t* __restrict__ xs) {
    __shared__ float sx[DDIM];
    __shared__ float sb[DDIM];
    const int b = blockIdx.y;
    const int tid = threadIdx.x;
    for (int j = tid; j < DDIM; j += 256) {
        sx[j] = x[b * DDIM + j];
        sb[j] = bsum[b * DDIM + j];
    }
    __syncthreads();
    const int lane = tid & 31;
    const int i = blockIdx.x * 8 + (tid >> 5);
    const float scal = (float)(DDIM - 1 - 2 * i);

    float mx = -3.402823466e38f;
    for (int j = lane; j < DDIM; j += 32) mx = fmaxf(mx, scal * sx[j] - sb[j]);
    #pragma unroll
    for (int off = 16; off > 0; off >>= 1) mx = fmaxf(mx, __shfl_xor(mx, off, 32));

    float se = 0.f, sw = 0.f;
    for (int j = lane; j < DDIM; j += 32) {
        const float e = __expf(scal * sx[j] - sb[j] - mx);
        se += e;
        sw += e * sx[j];
    }
    #pragma unroll
    for (int off = 16; off > 0; off >>= 1) {
        se += __shfl_xor(se, off, 32);
        sw += __shfl_xor(sw, off, 32);
    }
    if (lane == 0) xs[b * DDIM + i] = (bf16_t)(sw / se);
}

// ---------------------------------------------------------------------------
// Kernel W: fp32 -> bf16 weight conversion (once per launch; bandwidth-bound)
// ---------------------------------------------------------------------------
__global__ __launch_bounds__(256) void wconv_kernel(const float* __restrict__ w,
                                                    bf16_t* __restrict__ wb) {
    const int base = (blockIdx.x * 256 + threadIdx.x) * 4;
    const float4 v = *(const float4*)(w + base);
    v4bf o;
    o[0] = (bf16_t)v.x; o[1] = (bf16_t)v.y; o[2] = (bf16_t)v.z; o[3] = (bf16_t)v.w;
    *(v4bf*)(wb + base) = o;
}

// ---------------------------------------------------------------------------
// WMMA fragment loaders (all-bf16 path: pure b128 loads, no converts)
//   A 16x32 bf16: lane m=lr, elem  e    -> K=kk+hf*16+e
//   B 32x16 bf16: lane n=lr, elem  e    -> K=kk+hf*16+e
// ---------------------------------------------------------------------------
__device__ __forceinline__ v16bf load_afrag(const bf16_t* arow, int kk, int hf) {
    b32x8 t;
    t.a = *(const uint4*)(arow + kk + hf * 16);
    t.b = *(const uint4*)(arow + kk + hf * 16 + 8);
    return __builtin_bit_cast(v16bf, t);
}
__device__ __forceinline__ v16bf load_bfrag(const bf16_t* wrow, int kk, int hf) {
    b32x8 t;
    t.a = *(const uint4*)(wrow + kk + hf * 16);
    t.b = *(const uint4*)(wrow + kk + hf * 16 + 8);
    return __builtin_bit_cast(v16bf, t);
}

// ---------------------------------------------------------------------------
// Kernel C (primary): out_bf16 = leaky(A_bf16 @ Wb_bf16^T + bias)
// M=64, N=1024, K=1024; one 16x16 tile per wave; K unrolled x2 (2 WMMA/iter).
// ---------------------------------------------------------------------------
__global__ __launch_bounds__(256) void mlp_gemm_wmma_bf16(const bf16_t* __restrict__ A,
                                                          const bf16_t* __restrict__ Wb,
                                                          const float* __restrict__ bias,
                                                          bf16_t* __restrict__ out,
                                                          int apply_leaky) {
    const int lane = threadIdx.x & 31;
    const int wave = threadIdx.x >> 5;
    const int tile = blockIdx.x * 8 + wave;     // 0..255
    const int m0 = (tile & 3) << 4;
    const int n0 = (tile >> 2) << 4;
    const int hf = lane >> 4;
    const int lr = lane & 15;

    const bf16_t* arow = A  + (size_t)(m0 + lr) * DDIM;
    const bf16_t* wrow = Wb + (size_t)(n0 + lr) * DDIM;

    v8f c = {};
    for (int kk = 0; kk < DDIM; kk += 64) {
        v16bf a0 = load_afrag(arow, kk, hf);
        v16bf b0 = load_bfrag(wrow, kk, hf);
        v16bf a1 = load_afrag(arow, kk + 32, hf);
        v16bf b1 = load_bfrag(wrow, kk + 32, hf);
        c = __builtin_amdgcn_wmma_f32_16x16x32_bf16(false, a0, false, b0,
                                                    (short)0, c, false, false);
        c = __builtin_amdgcn_wmma_f32_16x16x32_bf16(false, a1, false, b1,
                                                    (short)0, c, false, false);
    }

    const int n = n0 + lr;
    const float bn = bias[n];
    #pragma unroll
    for (int r = 0; r < 8; ++r) {
        const int m = m0 + r + hf * 8;
        float v = c[r] + bn;
        if (apply_leaky) v = (v >= 0.f) ? v : NEG_SLOPE * v;
        out[(size_t)m * DDIM + n] = (bf16_t)v;
    }
}

// ---------------------------------------------------------------------------
// Kernel C (fallback, small-ws): fp32 weights converted in-register per step.
// ---------------------------------------------------------------------------
__global__ __launch_bounds__(256) void mlp_gemm_wmma_f32w(const bf16_t* __restrict__ A,
                                                          const float* __restrict__ W,
                                                          const float* __restrict__ bias,
                                                          bf16_t* __restrict__ out,
                                                          int apply_leaky) {
    const int lane = threadIdx.x & 31;
    const int wave = threadIdx.x >> 5;
    const int tile = blockIdx.x * 8 + wave;
    const int m0 = (tile & 3) << 4;
    const int n0 = (tile >> 2) << 4;
    const int hf = lane >> 4;
    const int lr = lane & 15;

    const bf16_t* arow = A + (size_t)(m0 + lr) * DDIM;
    const float*  wrow = W + (size_t)(n0 + lr) * DDIM;

    v8f c = {};
    for (int kk = 0; kk < DDIM; kk += 32) {
        v16bf af = load_afrag(arow, kk, hf);
        float bv[16];
        *(float4*)(bv + 0)  = *(const float4*)(wrow + kk + hf * 16);
        *(float4*)(bv + 4)  = *(const float4*)(wrow + kk + hf * 16 + 4);
        *(float4*)(bv + 8)  = *(const float4*)(wrow + kk + hf * 16 + 8);
        *(float4*)(bv + 12) = *(const float4*)(wrow + kk + hf * 16 + 12);
        v16bf wb;
        #pragma unroll
        for (int e = 0; e < 16; ++e) wb[e] = (bf16_t)bv[e];
        c = __builtin_amdgcn_wmma_f32_16x16x32_bf16(false, af, false, wb,
                                                    (short)0, c, false, false);
    }

    const int n = n0 + lr;
    const float bn = bias[n];
    #pragma unroll
    for (int r = 0; r < 8; ++r) {
        const int m = m0 + r + hf * 8;
        float v = c[r] + bn;
        if (apply_leaky) v = (v >= 0.f) ? v : NEG_SLOPE * v;
        out[(size_t)m * DDIM + n] = (bf16_t)v;
    }
}

// ---------------------------------------------------------------------------
// Kernel D: head  out[m,n] = h2[m,:] . W3[n,:] + b3[n],  n in {0,1}
// ---------------------------------------------------------------------------
__global__ __launch_bounds__(128) void head_kernel(const bf16_t* __restrict__ h,
                                                   const float* __restrict__ W3,
                                                   const float* __restrict__ b3,
                                                   float* __restrict__ out) {
    __shared__ float red[2][64];
    const int t = threadIdx.x;
    const int n = t >> 6, k0 = t & 63;
    const int m = blockIdx.x;
    float acc = 0.f;
    for (int k = k0; k < DDIM; k += 64)
        acc += (float)h[m * DDIM + k] * W3[n * DDIM + k];
    red[n][k0] = acc;
    __syncthreads();
    #pragma unroll
    for (int s = 32; s > 0; s >>= 1) {
        if (k0 < s) red[n][k0] += red[n][k0 + s];
        __syncthreads();
    }
    if (k0 == 0) out[m * 2 + n] = red[n][0] + b3[n];
}

// ---------------------------------------------------------------------------
extern "C" void kernel_launch(void* const* d_in, const int* in_sizes, int n_in,
                              void* d_out, int out_size, void* d_ws, size_t ws_size,
                              hipStream_t stream) {
    const float* x  = (const float*)d_in[0];
    const float* W1 = (const float*)d_in[1];
    const float* b1 = (const float*)d_in[2];
    const float* W2 = (const float*)d_in[3];
    const float* b2 = (const float*)d_in[4];
    const float* W3 = (const float*)d_in[5];
    const float* b3 = (const float*)d_in[6];
    float* out = (float*)d_out;

    // Workspace layout (all offsets 16B+ aligned)
    const size_t KB = 1024;
    unsigned char* w = (unsigned char*)d_ws;
    float*  bsum = (float*)(w);                    // 256 KB
    bf16_t* xs   = (bf16_t*)(w + 256 * KB);        // 128 KB
    bf16_t* h1   = (bf16_t*)(w + 384 * KB);        // 128 KB
    bf16_t* h2   = (bf16_t*)(w + 512 * KB);        // 128 KB
    bf16_t* W1b  = (bf16_t*)(w + 640 * KB);        // 2 MB
    bf16_t* W2b  = (bf16_t*)(w + 640 * KB + 2048 * KB);
    const size_t need_full = 640 * KB + 4096 * KB;
    const bool full = (ws_size >= need_full);      // constant per run -> deterministic

    bsum_kernel    <<<dim3(4, BATCH),   256, 0, stream>>>(x, bsum);
    softsort_kernel<<<dim3(128, BATCH), 256, 0, stream>>>(x, bsum, xs);

    if (full) {
        wconv_kernel<<<1024, 256, 0, stream>>>(W1, W1b);
        wconv_kernel<<<1024, 256, 0, stream>>>(W2, W2b);
        mlp_gemm_wmma_bf16<<<32, 256, 0, stream>>>(xs, W1b, b1, h1, 1);
        mlp_gemm_wmma_bf16<<<32, 256, 0, stream>>>(h1, W2b, b2, h2, 1);
    } else {
        mlp_gemm_wmma_f32w<<<32, 256, 0, stream>>>(xs, W1, b1, h1, 1);
        mlp_gemm_wmma_f32w<<<32, 256, 0, stream>>>(h1, W2, b2, h2, 1);
    }
    head_kernel<<<BATCH, 128, 0, stream>>>(h2, W3, b3, out);
}
